// SelfAttention2varible_8555574854295
// MI455X (gfx1250) — matrix operand order repaired and checked by reading, not compiled, not run on hardware
//
#include <hip/hip_runtime.h>
#include <hip/hip_bf16.h>
#include <math.h>

// ---------------------------------------------------------------------------
// SelfAttention (B=8, C=256, H=W=32, 8 heads, d=32, GroupNorm(32 groups))
// f16 WMMA pipeline: groupnorm -> proj GEMMs (64x16 wave tiles) ->
// flash attention (K/V staged in LDS via TDM tensor_load_to_lds) -> out GEMM
// ---------------------------------------------------------------------------

typedef __attribute__((ext_vector_type(16))) _Float16 v16h;
typedef __attribute__((ext_vector_type(8)))  _Float16 v8h;
typedef __attribute__((ext_vector_type(8)))  float    v8f;
typedef __attribute__((ext_vector_type(4)))  unsigned int u32x4;
typedef __attribute__((ext_vector_type(8)))  int i32x8;
typedef __attribute__((ext_vector_type(4)))  int i32x4;

union FragA { v16h v; v8h h[2]; };

#define WMMA_F16(A, Bm, Cv) \
  __builtin_amdgcn_wmma_f32_16x16x32_f16(false, (A), false, (Bm), (short)0, (Cv), false, false)

// ---------------- TDM: 1-D contiguous global -> LDS DMA (D# per ISA 8.3/8.4)
// data_size = 8 bytes; bytes must be a multiple of 8 and <= 64KB*8.
__device__ __forceinline__ void tdm_load_1d(unsigned lds_off, const void* gptr,
                                            unsigned bytes) {
  const unsigned long long ga = (unsigned long long)gptr;
  const unsigned long long n8 = bytes >> 3;  // 8-byte units (tile_dim0, 16-bit)

  union { unsigned int d[4]; u32x4 v; } g0;
  g0.d[0] = 1u;                                            // count=1, user mode
  g0.d[1] = lds_off;                                       // lds_addr
  g0.d[2] = (unsigned)(ga & 0xffffffffull);                // global_addr[31:0]
  g0.d[3] = (unsigned)((ga >> 32) & 0x1ffffffull) | (2u << 30);  // ga[56:32], type=2

  union { unsigned long long q[4]; i32x8 v; } g1;
  g1.q[0] = (3ull << 16)                                   // data_size = 8B
          | ((n8 & 0xffffull) << 48);                      // tensor_dim0[15:0]
  g1.q[1] = (n8 >> 16)                                     // tensor_dim0[31:16]
          | (1ull << 16)                                   // tensor_dim1 = 1
          | ((n8 & 0xffffull) << 48);                      // tile_dim0
  g1.q[2] = 1ull                                           // tile_dim1 = 1
          | (n8 << 32);                                    // tensor_dim0_stride
  g1.q[3] = (n8 << 16);                                    // tensor_dim1_stride

  i32x4 z4 = {0, 0, 0, 0};
#if __clang_major__ >= 23
  i32x8 z8 = {0, 0, 0, 0, 0, 0, 0, 0};
  __builtin_amdgcn_tensor_load_to_lds(g0.v, g1.v, z4, z4, z8, 0);
#else
  __builtin_amdgcn_tensor_load_to_lds(g0.v, g1.v, z4, z4, 0);
#endif
}

// -------------------------------- f32 -> f16 --------------------------------
__global__ void f32_to_f16_kernel(const float* __restrict__ src,
                                  _Float16* __restrict__ dst, int n) {
  int i = blockIdx.x * 256 + threadIdx.x;
  if (i < n) dst[i] = (_Float16)src[i];
}

// ------------------------- GroupNorm (transposed out) -----------------------
__global__ void groupnorm_kernel(const float* __restrict__ xq,
                                 const float* __restrict__ xkv,
                                 const float* __restrict__ sq,
                                 const float* __restrict__ bq,
                                 const float* __restrict__ skv,
                                 const float* __restrict__ bkv,
                                 _Float16* __restrict__ xnq,
                                 _Float16* __restrict__ xnkv) {
  const int g = blockIdx.x, b = blockIdx.y, which = blockIdx.z;
  const float* x  = which ? xkv : xq;
  const float* sc = which ? skv : sq;
  const float* bi = which ? bkv : bq;
  _Float16* out   = which ? xnkv : xnq;
  const int tid = threadIdx.x;
  const float* xg = x + ((size_t)b * 256 + g * 8) * 1024;  // 8 ch x 1024 sp

  float s = 0.f, s2 = 0.f;
  for (int i = tid; i < 8192; i += 256) { float v = xg[i]; s += v; s2 += v * v; }
  __shared__ float r1[256], r2[256];
  r1[tid] = s; r2[tid] = s2; __syncthreads();
  for (int off = 128; off > 0; off >>= 1) {
    if (tid < off) { r1[tid] += r1[tid + off]; r2[tid] += r2[tid + off]; }
    __syncthreads();
  }
  const float mean = r1[0] * (1.f / 8192.f);
  const float var  = r2[0] * (1.f / 8192.f) - mean * mean;
  const float rstd = rsqrtf(var + 1e-5f);

  for (int i = tid; i < 8192; i += 256) {
    int j = i >> 10, spos = i & 1023;
    int c = g * 8 + j;
    float v = (xg[i] - mean) * rstd * sc[c] + bi[c];
    out[((size_t)b * 1024 + spos) * 256 + c] = (_Float16)v;
  }
}

// ------------------------------ wave-tile GEMM ------------------------------
// A:[M][K] f16 row-major; Bt per-batch [N][K] f16. 64(M)x16(N) per wave.
__global__ void __launch_bounds__(32)
gemm64_kernel(const _Float16* __restrict__ A,
              const _Float16* __restrict__ Bt,
              int N, int K, int mode,
              _Float16* __restrict__ outA,
              _Float16* __restrict__ outB,
              const float* __restrict__ bout,
              const float* __restrict__ resid,
              float* __restrict__ outF) {
  const int lane = threadIdx.x;
  const int g  = lane >> 4;
  const int lm = lane & 15;
  const int n0 = blockIdx.x * 16;
  const int m0 = blockIdx.y * 64;
  const int b  = blockIdx.z;

  const _Float16* Arow = A + (size_t)(m0 + lm) * K;
  const _Float16* Brow = Bt + (size_t)b * N * K + (size_t)(n0 + lm) * K;

  __builtin_prefetch(Brow, 0, 0);
  __builtin_prefetch(Arow, 0, 0);

  v8f acc[4] = {{}, {}, {}, {}};
  for (int k0 = 0; k0 < K; k0 += 32) {
    FragA fb;
    fb.h[0] = *(const v8h*)(Brow + k0 + 8 * g);
    fb.h[1] = *(const v8h*)(Brow + k0 + 8 * g + 16);
#pragma unroll
    for (int t = 0; t < 4; ++t) {
      FragA fa;
      const _Float16* ar = Arow + (size_t)t * 16 * K + k0;
      fa.h[0] = *(const v8h*)(ar + 8 * g);
      fa.h[1] = *(const v8h*)(ar + 8 * g + 16);
      acc[t] = WMMA_F16(fa.v, fb.v, acc[t]);
    }
  }

  const int n = n0 + lm;
#pragma unroll
  for (int t = 0; t < 4; ++t) {
    const int mbase = m0 + t * 16 + 8 * g;
    if (mode == 2) {
#pragma unroll
      for (int r = 0; r < 8; ++r) {
        size_t idx = ((size_t)b * 256 + mbase + r) * 1024 + n;
        outF[idx] = acc[t][r] + bout[mbase + r] + resid[idx];
      }
    } else if (mode == 1 && mbase >= 256) {
      const int o = mbase - 256, nh = o >> 5, ddb = o & 31;
#pragma unroll
      for (int r = 0; r < 8; ++r)
        outB[(((size_t)b * 8 + nh) * 32 + ddb + r) * 1024 + n] = (_Float16)acc[t][r];
    } else {  // packed [s][d] store, dd multiple of 8
      const int nh = mbase >> 5, dd = mbase & 31;
      v8h pk;
#pragma unroll
      for (int r = 0; r < 8; ++r) pk[r] = (_Float16)acc[t][r];
      *(v8h*)(outA + (((size_t)b * 8 + nh) * 1024 + n) * 32 + dd) = pk;
    }
  }
}

// ------------------------------ flash attention -----------------------------
// One 8-wave workgroup per (b, head). Wave 0 DMAs the whole K (64KB) and V
// (64KB) into LDS via TDM; s_wait_tensorcnt + barrier; then each wave runs
// flash attention on 4 s-blocks of 32 rows, streaming K/V from LDS.
__global__ void __launch_bounds__(256)
attn_kernel(const _Float16* __restrict__ qT,
            const _Float16* __restrict__ kT,
            const _Float16* __restrict__ vv,
            _Float16* __restrict__ attnout) {
  __shared__ __align__(128) _Float16 Klds[1024 * 32];  // [t][d]  64KB
  __shared__ __align__(128) _Float16 Vlds[32 * 1024];  // [dd][t] 64KB
  __shared__ __align__(64)  _Float16 P[8][16][32];     // per-wave P staging 8KB

  const int nh = blockIdx.x, b = blockIdx.y;
  const int tid  = threadIdx.x;
  const int wid  = tid >> 5, lane = tid & 31;
  const int g = lane >> 4, lm = lane & 15;

  const _Float16* qbase = qT + ((size_t)b * 8 + nh) * 1024 * 32;
  const _Float16* kbase = kT + ((size_t)b * 8 + nh) * 1024 * 32;
  const _Float16* vbase = vv + ((size_t)b * 8 + nh) * 32 * 1024;

  if (wid == 0) {
    tdm_load_1d((unsigned)(size_t)&Klds[0], kbase, 1024 * 32 * 2);
    tdm_load_1d((unsigned)(size_t)&Vlds[0], vbase, 32 * 1024 * 2);
    __builtin_amdgcn_s_wait_tensorcnt(0);
  }
  __syncthreads();

  const float scl = 1.0f / 16.0f;  // 1/sqrt(C)

  for (int sb = 0; sb < 4; ++sb) {
    const int s0 = wid * 128 + sb * 32;

    FragA qa[2];
#pragma unroll
    for (int u = 0; u < 2; ++u) {
      const _Float16* qrow = qbase + (size_t)(s0 + u * 16 + lm) * 32;
      qa[u].h[0] = *(const v8h*)(qrow + 8 * g);
      qa[u].h[1] = *(const v8h*)(qrow + 8 * g + 16);
    }

    float rowm[2][8], rowl[2][8];
    v8f o0[2] = {{}, {}}, o1[2] = {{}, {}};
#pragma unroll
    for (int u = 0; u < 2; ++u)
#pragma unroll
      for (int r = 0; r < 8; ++r) { rowm[u][r] = -1e30f; rowl[u][r] = 0.f; }

    for (int t0 = 0; t0 < 1024; t0 += 32) {
      FragA kb0, kb1, vb0, vb1;                       // ds_load_b128 pairs
      kb0.h[0] = *(const v8h*)(Klds + (size_t)(t0 + lm) * 32 + 8 * g);
      kb0.h[1] = *(const v8h*)(Klds + (size_t)(t0 + lm) * 32 + 8 * g + 16);
      kb1.h[0] = *(const v8h*)(Klds + (size_t)(t0 + 16 + lm) * 32 + 8 * g);
      kb1.h[1] = *(const v8h*)(Klds + (size_t)(t0 + 16 + lm) * 32 + 8 * g + 16);
      vb0.h[0] = *(const v8h*)(Vlds + (size_t)lm * 1024 + t0 + 8 * g);
      vb0.h[1] = *(const v8h*)(Vlds + (size_t)lm * 1024 + t0 + 8 * g + 16);
      vb1.h[0] = *(const v8h*)(Vlds + (size_t)(16 + lm) * 1024 + t0 + 8 * g);
      vb1.h[1] = *(const v8h*)(Vlds + (size_t)(16 + lm) * 1024 + t0 + 8 * g + 16);

#pragma unroll
      for (int u = 0; u < 2; ++u) {
        v8f sa = {}, sb2 = {};
        sa  = WMMA_F16(qa[u].v, kb0.v, sa);    // S cols t0..t0+15
        sb2 = WMMA_F16(qa[u].v, kb1.v, sb2);   // S cols t0+16..t0+31

#pragma unroll
        for (int r = 0; r < 8; ++r) {
          float x0 = sa[r] * scl, x1 = sb2[r] * scl;
          float mx = fmaxf(x0, x1);
          for (int off = 8; off > 0; off >>= 1)
            mx = fmaxf(mx, __shfl_xor(mx, off, 32));   // within 16-lane half
          float newm = fmaxf(rowm[u][r], mx);
          float corr = __expf(rowm[u][r] - newm);
          float p0 = __expf(x0 - newm);
          float p1 = __expf(x1 - newm);
          float ls = p0 + p1;
          for (int off = 8; off > 0; off >>= 1)
            ls += __shfl_xor(ls, off, 32);
          rowl[u][r] = rowl[u][r] * corr + ls;
          rowm[u][r] = newm;
          o0[u][r] *= corr; o1[u][r] *= corr;
          const int m = r + 8 * g;
          P[wid][m][lm]      = (_Float16)p0;
          P[wid][m][16 + lm] = (_Float16)p1;
        }
        // same-wave LDS ops are in-order (DScnt rules); fence the compiler only
        asm volatile("" ::: "memory");
        FragA pa;
        pa.h[0] = *(const v8h*)(&P[wid][lm][8 * g]);
        pa.h[1] = *(const v8h*)(&P[wid][lm][8 * g + 16]);
        asm volatile("" ::: "memory");

        o0[u] = WMMA_F16(pa.v, vb0.v, o0[u]);   // dd = 0..15
        o1[u] = WMMA_F16(pa.v, vb1.v, o1[u]);   // dd = 16..31
      }
    }

#pragma unroll
    for (int u = 0; u < 2; ++u)
#pragma unroll
      for (int r = 0; r < 8; ++r) {
        const int s = s0 + u * 16 + r + 8 * g;
        const float inv = 1.f / rowl[u][r];
        const size_t base = ((size_t)b * 1024 + s) * 256;
        attnout[base + nh * 32 + lm]      = (_Float16)(o0[u][r] * inv);
        attnout[base + nh * 32 + 16 + lm] = (_Float16)(o1[u][r] * inv);
      }
  }
}

// --------------------------------- launcher ---------------------------------
extern "C" void kernel_launch(void* const* d_in, const int* in_sizes, int n_in,
                              void* d_out, int out_size, void* d_ws, size_t ws_size,
                              hipStream_t stream) {
  (void)in_sizes; (void)n_in; (void)out_size; (void)ws_size;
  const float* input_q   = (const float*)d_in[0];
  const float* input_kv  = (const float*)d_in[1];
  const float* gq_scale  = (const float*)d_in[2];
  const float* gq_bias   = (const float*)d_in[3];
  const float* gkv_scale = (const float*)d_in[4];
  const float* gkv_bias  = (const float*)d_in[5];
  const float* Wq        = (const float*)d_in[6];
  const float* Wkv       = (const float*)d_in[7];
  const float* Wout      = (const float*)d_in[8];
  const float* bout      = (const float*)d_in[9];
  float* out = (float*)d_out;

  char* ws = (char*)d_ws;
  _Float16* xnq     = (_Float16*)(ws);                                // 4 MB
  _Float16* xnkv    = (_Float16*)(ws + (4ull << 20));                 // 4 MB
  _Float16* wq_h    = (_Float16*)(ws + (8ull << 20));                 // 128 KB
  _Float16* wkv_h   = (_Float16*)(ws + (8ull << 20) + (128ull << 10));// 256 KB
  _Float16* wout_h  = (_Float16*)(ws + (8ull << 20) + (384ull << 10));// 128 KB
  _Float16* qT      = (_Float16*)(ws + (8ull << 20) + (512ull << 10));// 4 MB
  _Float16* kT      = (_Float16*)(ws + (12ull << 20) + (512ull << 10));// 4 MB
  _Float16* vv      = (_Float16*)(ws + (16ull << 20) + (512ull << 10));// 4 MB
  _Float16* attnout = (_Float16*)(ws + (20ull << 20) + (512ull << 10));// 4 MB

  f32_to_f16_kernel<<<65536 / 256, 256, 0, stream>>>(Wq, wq_h, 65536);
  f32_to_f16_kernel<<<131072 / 256, 256, 0, stream>>>(Wkv, wkv_h, 131072);
  f32_to_f16_kernel<<<65536 / 256, 256, 0, stream>>>(Wout, wout_h, 65536);

  groupnorm_kernel<<<dim3(32, 8, 2), 256, 0, stream>>>(
      input_q, input_kv, gq_scale, gq_bias, gkv_scale, gkv_bias, xnq, xnkv);

  // query = Wq @ norm_q  -> qT[b][nh][s][d]
  gemm64_kernel<<<dim3(64, 4, 8), 32, 0, stream>>>(
      wq_h, xnq, 1024, 256, 0, qT, nullptr, nullptr, nullptr, nullptr);
  // kv = Wkv @ norm_kv  -> kT[b][nh][s][d], v[b][nh][d][t]
  gemm64_kernel<<<dim3(64, 8, 8), 32, 0, stream>>>(
      wkv_h, xnkv, 1024, 256, 1, kT, vv, nullptr, nullptr, nullptr);

  // attention: one workgroup per (head, batch), K/V staged in LDS via TDM
  attn_kernel<<<dim3(8, 8), 256, 0, stream>>>(qT, kT, vv, attnout);

  // out = Wout @ attn_out + bout + input_q
  gemm64_kernel<<<dim3(64, 4, 8), 32, 0, stream>>>(
      wout_h, attnout, 1024, 256, 2, nullptr, nullptr, bout, input_q, out);
}
